// DoubleStreamBlock_40252433498505
// MI455X (gfx1250) — matrix operand order repaired and checked by reading, not compiled, not run on hardware
//
#include <hip/hip_runtime.h>
#include <math.h>

typedef __attribute__((ext_vector_type(16))) _Float16 v16h;
typedef __attribute__((ext_vector_type(2)))  _Float16 h2_t;
typedef __attribute__((ext_vector_type(8)))  float    v8f;
typedef __attribute__((ext_vector_type(4)))  unsigned int u32x4;
typedef __attribute__((ext_vector_type(8)))  int      i32x8;
typedef __attribute__((ext_vector_type(4)))  int      i32x4;

#define LT    256
#define LI    2048
#define LTOT  2304
#define DIM   2048
#define NH    16
#define HDIM  128
#define MLPD  8192

// K-index mapping of the 16-bit A-matrix WMMA fragment (CDNA5 ISA):
// lane group g = lane/16, VGPR v holds K pair: (v<4 ? 2v : 16+2(v-4)) + 8g
__device__ __forceinline__ int kmap(int v, int g) {
  return (v < 4 ? 2 * v : 16 + 2 * (v - 4)) + 8 * g;
}

__device__ __forceinline__ v8f wmma16x16x32(v16h a, v16h b, v8f c) {
  return __builtin_amdgcn_wmma_f32_16x16x32_f16(false, a, false, b, (short)0, c,
                                                false, false);
}

// ---------------------------------------------------------------------------
// TDM: issue a contiguous/strided 2D tile load Global -> LDS (D# per ISA 8.3/8.4)
// data_size = 2 bytes. elems = tile_dim0 (<= 65535), rows = tile_dim1,
// stride = tensor_dim0_stride in elements.
// 6-arg builtin form (clang-23 / therock-10.0 lane).
// ---------------------------------------------------------------------------
__device__ __forceinline__ void tdm_load_2d_f16(unsigned int lds_off,
                                                const void* gptr,
                                                unsigned int elems,
                                                unsigned int rows,
                                                unsigned int stride) {
  unsigned long long ga = (unsigned long long)(uintptr_t)gptr;
  u32x4 g0;
  g0[0] = 1u;                                   // count=1, user descriptor
  g0[1] = lds_off;                              // lds_addr (bytes)
  g0[2] = (unsigned int)ga;                     // global_addr[31:0]
  g0[3] = (unsigned int)((ga >> 32) & 0x01FFFFFFu) | (2u << 30);  // ga[56:32]|type=2
  i32x8 g1;
  g1[0] = (int)(1u << 16);                      // workgroup_mask=0, data_size=1 (2B)
  g1[1] = (int)((elems & 0xFFFFu) << 16);       // tensor_dim0[15:0] (atomic_addr=0)
  g1[2] = (int)(((elems >> 16) & 0xFFFFu) | ((rows & 0xFFFFu) << 16)); // td0 hi|td1 lo
  g1[3] = (int)(((rows >> 16) & 0xFFFFu) | ((elems & 0xFFFFu) << 16)); // td1 hi|tile_dim0
  g1[4] = (int)(rows & 0xFFFFu);                // tile_dim1 (tile_dim2=0)
  g1[5] = (int)stride;                          // tensor_dim0_stride[31:0]
  g1[6] = 0;                                    // stride hi | tensor_dim1_stride lo
  g1[7] = 0;
  i32x4 gz;
  gz[0] = 0; gz[1] = 0; gz[2] = 0; gz[3] = 0;   // groups 2/3 unused (<=2D)
  i32x8 gz8;
  gz8[0] = 0; gz8[1] = 0; gz8[2] = 0; gz8[3] = 0;
  gz8[4] = 0; gz8[5] = 0; gz8[6] = 0; gz8[7] = 0;
  __builtin_amdgcn_tensor_load_to_lds(g0, g1, gz, gz, gz8, 0);
}

// ---------------------------------------------------------------------------
// Stage 0: mod = silu(vec) @ mod_w + mod_b   (two streams, 6*2048 each)
// mod layout: [stream(0=img,1=txt)][chunk 0..5][2048]
// ---------------------------------------------------------------------------
__global__ __launch_bounds__(256)
void k_modulation(const float* __restrict__ vec,
                  const float* __restrict__ wI, const float* __restrict__ bI,
                  const float* __restrict__ wT, const float* __restrict__ bT,
                  float* __restrict__ mod) {
  __shared__ float sv[256];
  int j  = blockIdx.x * 256 + threadIdx.x;     // 0..24575
  int s  = j / (6 * DIM);                      // 0=img, 1=txt
  int jj = j % (6 * DIM);
  const float* W  = s ? wT : wI;
  const float* Bb = s ? bT : bI;
  float acc = 0.f;
  for (int d0 = 0; d0 < DIM; d0 += 256) {
    float x = vec[d0 + threadIdx.x];
    sv[threadIdx.x] = x / (1.f + __expf(-x));  // silu
    __syncthreads();
#pragma unroll 8
    for (int d = 0; d < 256; ++d)
      acc += sv[d] * W[(size_t)(d0 + d) * (6 * DIM) + jj];
    __syncthreads();
  }
  mod[(size_t)s * (6 * DIM) + jj] = acc + Bb[jj];
}

// ---------------------------------------------------------------------------
// LayerNorm + modulate: out[t] = (1+sc)*LN(x[t]) + sh   (f16 output)
// token t<LT -> txt row t, else img row t-LT
// ---------------------------------------------------------------------------
__global__ __launch_bounds__(256)
void k_ln_mod(const float* __restrict__ imgSrc, const float* __restrict__ txtSrc,
              const float* __restrict__ mod, int sh_idx, int sc_idx,
              _Float16* __restrict__ out) {
  int t = blockIdx.x;
  const float* x = (t < LT) ? (txtSrc + (size_t)t * DIM)
                            : (imgSrc + (size_t)(t - LT) * DIM);
  const float* m  = (t < LT) ? (mod + 6 * DIM) : mod;
  const float* sh = m + sh_idx * DIM;
  const float* sc = m + sc_idx * DIM;
  int tid = threadIdx.x, lane = tid & 31, wid = tid >> 5;
  float v[8], s1 = 0.f, s2 = 0.f;
#pragma unroll
  for (int i = 0; i < 8; ++i) {
    v[i] = x[tid + i * 256];
    s1 += v[i];
    s2 += v[i] * v[i];
  }
#pragma unroll
  for (int o = 16; o > 0; o >>= 1) {
    s1 += __shfl_xor(s1, o, 32);
    s2 += __shfl_xor(s2, o, 32);
  }
  __shared__ float r1[8], r2[8];
  if (lane == 0) { r1[wid] = s1; r2[wid] = s2; }
  __syncthreads();
  if (wid == 0) {
    s1 = (lane < 8) ? r1[lane] : 0.f;
    s2 = (lane < 8) ? r2[lane] : 0.f;
#pragma unroll
    for (int o = 4; o > 0; o >>= 1) {
      s1 += __shfl_xor(s1, o, 32);
      s2 += __shfl_xor(s2, o, 32);
    }
    if (lane == 0) { r1[0] = s1; r2[0] = s2; }
  }
  __syncthreads();
  float mu   = r1[0] * (1.f / DIM);
  float var  = r2[0] * (1.f / DIM) - mu * mu;
  float rstd = rsqrtf(var + 1e-6f);
#pragma unroll
  for (int i = 0; i < 8; ++i) {
    int c = tid + i * 256;
    float y = (v[i] - mu) * rstd;
    y = (1.f + sc[c]) * y + sh[c];
    out[(size_t)t * DIM + c] = (_Float16)y;
  }
}

// ---------------------------------------------------------------------------
// WMMA GEMM: C[M,N] = A[M,K](f16) * B[K,N](f32->f16) (+bias) (opt GELU)
// BM=128 BN=128 BK=32; 256 threads = 8 waves arranged 2x4.
// A staged into LDS pre-swizzled in WMMA A-fragment order -> contiguous reads.
// ---------------------------------------------------------------------------
template <int OUT_F16, int BIAS, int GELU>
__global__ __launch_bounds__(256)
void k_gemm(const _Float16* __restrict__ A, int lda,
            const float* __restrict__ B, int ldb,
            const float* __restrict__ bias,
            void* __restrict__ Cout, int ldc,
            int M, int N, int K) {
  __shared__ v16h As[8][32];                        // 8 KB, swizzled frags
  __shared__ __align__(32) _Float16 Bs[32][128];    // 8 KB, row-major [k][n]
  const int tid  = threadIdx.x;
  const int lane = tid & 31, wid = tid >> 5;
  const int g = (lane >> 4) & 1, l16 = lane & 15;
  const int wrow = wid >> 2, wcol = wid & 3;        // 2 x 4 wave grid
  const int m0 = blockIdx.y * 128, n0 = blockIdx.x * 128;

  v8f acc[4][2];
#pragma unroll
  for (int mi = 0; mi < 4; ++mi)
#pragma unroll
    for (int nj = 0; nj < 2; ++nj)
      acc[mi][nj] = v8f{0.f, 0.f, 0.f, 0.f, 0.f, 0.f, 0.f, 0.f};

  for (int k0 = 0; k0 < K; k0 += 32) {
    // --- stage A tile (pre-swizzled into fragment layout) ---
    {
      int afr = tid >> 5, al = tid & 31;
      int ag = (al >> 4) & 1, al16 = al & 15;
      const _Float16* Ap = A + (size_t)(m0 + afr * 16 + al16) * lda + k0;
      v16h av;
#pragma unroll
      for (int v = 0; v < 8; ++v) {
        h2_t p = *(const h2_t*)(Ap + kmap(v, ag));
        av[2 * v]     = p[0];
        av[2 * v + 1] = p[1];
      }
      As[afr][al] = av;
    }
    // --- stage B tile (convert fp32 -> f16) ---
    {
      int kr = tid >> 3, seg = tid & 7;
      const float* Bp = B + (size_t)(k0 + kr) * ldb + n0 + seg * 16;
#pragma unroll
      for (int q = 0; q < 16; ++q)
        Bs[kr][seg * 16 + q] = (_Float16)Bp[q];
    }
    __syncthreads();

    if (k0 + 32 < K) {  // prefetch next tiles -> global_prefetch_b8
      __builtin_prefetch(A + (size_t)(m0 + (tid >> 1)) * lda + k0 + 32, 0, 1);
      __builtin_prefetch(B + (size_t)(k0 + 32 + lane) * ldb + n0, 0, 1);
    }

    v16h bfr0, bfr1;
#pragma unroll
    for (int s = 0; s < 16; ++s) {
      int kk = kmap(s >> 1, g) + (s & 1);
      bfr0[s] = Bs[kk][wcol * 32 + l16];
      bfr1[s] = Bs[kk][wcol * 32 + 16 + l16];
    }
#pragma unroll
    for (int mi = 0; mi < 4; ++mi) {
      v16h afr = As[wrow * 4 + mi][lane];
      acc[mi][0] = wmma16x16x32(afr, bfr0, acc[mi][0]);
      acc[mi][1] = wmma16x16x32(afr, bfr1, acc[mi][1]);
    }
    __syncthreads();
  }

  // --- epilogue: C layout row = mi*16 + v + 8g, col = nj*16 + l16 ---
#pragma unroll
  for (int mi = 0; mi < 4; ++mi) {
    int rowb = m0 + wrow * 64 + mi * 16 + g * 8;
#pragma unroll
    for (int nj = 0; nj < 2; ++nj) {
      int col = n0 + wcol * 32 + nj * 16 + l16;
      float bb = BIAS ? bias[col] : 0.f;
#pragma unroll
      for (int v = 0; v < 8; ++v) {
        float val = acc[mi][nj][v] + bb;
        if (GELU) {
          float u = val;
          val = 0.5f * u * (1.f + tanhf(0.7978845608f * (u + 0.044715f * u * u * u)));
        }
        if (OUT_F16)
          ((_Float16*)Cout)[(size_t)(rowb + v) * ldc + col] = (_Float16)val;
        else
          ((float*)Cout)[(size_t)(rowb + v) * ldc + col] = val;
      }
    }
  }
}

// ---------------------------------------------------------------------------
// RMSNorm(q,k) + RoPE, split heads -> f16 Q/K/V [NH][LTOT][HDIM]
// qkv fp32 layout: [t][s(3)][h(16)][d(128)]
// ---------------------------------------------------------------------------
__global__ __launch_bounds__(64)
void k_qknorm_rope(const float* __restrict__ qkv, const float* __restrict__ pe,
                   const float* __restrict__ qsI, const float* __restrict__ ksI,
                   const float* __restrict__ qsT, const float* __restrict__ ksT,
                   _Float16* __restrict__ Q, _Float16* __restrict__ K,
                   _Float16* __restrict__ V) {
  int h = blockIdx.x, t = blockIdx.y;
  int d = threadIdx.x;  // pair (2d, 2d+1)
  const float* base = qkv + (size_t)t * (3 * DIM) + h * HDIM;
  float q0 = base[2 * d], q1 = base[2 * d + 1];
  float k0 = base[DIM + 2 * d], k1 = base[DIM + 2 * d + 1];
  float v0 = base[2 * DIM + 2 * d], v1 = base[2 * DIM + 2 * d + 1];
  float ssq = q0 * q0 + q1 * q1, ssk = k0 * k0 + k1 * k1;
#pragma unroll
  for (int o = 16; o > 0; o >>= 1) {
    ssq += __shfl_xor(ssq, o, 32);
    ssk += __shfl_xor(ssk, o, 32);
  }
  __shared__ float sq[2], sk[2];
  int wid = threadIdx.x >> 5, lane = threadIdx.x & 31;
  if (lane == 0) { sq[wid] = ssq; sk[wid] = ssk; }
  __syncthreads();
  float rq = rsqrtf((sq[0] + sq[1]) * (1.f / HDIM) + 1e-6f);
  float rk = rsqrtf((sk[0] + sk[1]) * (1.f / HDIM) + 1e-6f);
  const float* qs = (t < LT) ? qsT : qsI;
  const float* ks = (t < LT) ? ksT : ksI;
  q0 = q0 * rq * qs[2 * d]; q1 = q1 * rq * qs[2 * d + 1];
  k0 = k0 * rk * ks[2 * d]; k1 = k1 * rk * ks[2 * d + 1];
  const float* p = pe + ((size_t)t * 64 + d) * 4;  // [i][j][r]
  float pq0 = p[0] * q0 + p[1] * q1;
  float pq1 = p[2] * q0 + p[3] * q1;
  float pk0 = p[0] * k0 + p[1] * k1;
  float pk1 = p[2] * k0 + p[3] * k1;
  size_t o = ((size_t)h * LTOT + t) * HDIM + 2 * d;
  Q[o] = (_Float16)pq0; Q[o + 1] = (_Float16)pq1;
  K[o] = (_Float16)pk0; K[o + 1] = (_Float16)pk1;
  V[o] = (_Float16)v0;  V[o + 1] = (_Float16)v1;
}

// ---------------------------------------------------------------------------
// Flash attention with WMMA. Block = (head, 128 q rows), 8 waves x 16 rows.
// V tile staged by the Tensor Data Mover (tensor_load_to_lds / TENSORcnt),
// K tile staged transposed by the VALU path for the K^T B-fragments.
// ---------------------------------------------------------------------------
__global__ __launch_bounds__(256)
void k_attn(const _Float16* __restrict__ Q, const _Float16* __restrict__ K,
            const _Float16* __restrict__ V, _Float16* __restrict__ O) {
  const int h = blockIdx.x, bq = blockIdx.y;
  const int tid = threadIdx.x, lane = tid & 31, wid = tid >> 5;
  const int g = (lane >> 4) & 1, l16 = lane & 15;
  const _Float16* Qh = Q + (size_t)h * LTOT * HDIM;
  const _Float16* Kh = K + (size_t)h * LTOT * HDIM;
  const _Float16* Vh = V + (size_t)h * LTOT * HDIM;

  __shared__ __align__(32) _Float16 Kt[HDIM][32];   // K^T tile [d][kv] 8 KB
  __shared__ __align__(32) _Float16 Vt[32][HDIM];   // V tile   [kv][d] 8 KB (TDM dst)
  __shared__ __align__(32) _Float16 Ps[8][16][32];  // P per wave        8 KB

  // Q fragments for this wave's 16 rows (A-layout)
  v16h qf[4];
  {
    const _Float16* Qp = Qh + (size_t)(bq * 128 + wid * 16 + l16) * HDIM;
#pragma unroll
    for (int kc = 0; kc < 4; ++kc) {
      v16h a;
#pragma unroll
      for (int v = 0; v < 8; ++v) {
        h2_t p = *(const h2_t*)(Qp + kc * 32 + kmap(v, g));
        a[2 * v] = p[0]; a[2 * v + 1] = p[1];
      }
      qf[kc] = a;
    }
  }

  const unsigned int ldsV = (unsigned int)(uintptr_t)&Vt[0][0];  // LDS byte offset

  v8f o8[8];
#pragma unroll
  for (int n = 0; n < 8; ++n) o8[n] = v8f{0.f,0.f,0.f,0.f,0.f,0.f,0.f,0.f};
  float rowmax[8], rowsum[8];
#pragma unroll
  for (int v = 0; v < 8; ++v) { rowmax[v] = -1e30f; rowsum[v] = 0.f; }

  const float scale = 0.08838834764831845f;  // 1/sqrt(128)

  for (int kv0 = 0; kv0 < LTOT; kv0 += 32) {
    __syncthreads();  // previous tile fully consumed
    // --- wave 0: kick the TDM for the V tile (32x128 f16, contiguous 8KB) ---
    if (wid == 0) {
      tdm_load_2d_f16(ldsV, Vh + (size_t)kv0 * HDIM, 32 * HDIM, 1, 32 * HDIM);
    }
    // --- all waves: stage K transposed for the K^T fragments ---
    {
      int kv = tid >> 3, seg = tid & 7;
      const _Float16* Kp = Kh + (size_t)(kv0 + kv) * HDIM + seg * 16;
#pragma unroll
      for (int q = 0; q < 16; ++q) Kt[seg * 16 + q][kv] = Kp[q];
      if (kv0 + 32 < LTOT)  // prefetch next K tile -> global_prefetch_b8
        __builtin_prefetch(Kh + (size_t)(kv0 + 32 + kv) * HDIM + seg * 16, 0, 1);
    }
    if (wid == 0) __builtin_amdgcn_s_wait_tensorcnt(0);  // V tile landed
    __syncthreads();

    // S = Q * K^T  (two 16x16 C tiles over 32 kv)
    v8f s0 = v8f{0.f,0.f,0.f,0.f,0.f,0.f,0.f,0.f};
    v8f s1 = v8f{0.f,0.f,0.f,0.f,0.f,0.f,0.f,0.f};
#pragma unroll
    for (int kc = 0; kc < 4; ++kc) {
      v16h b0, b1;
#pragma unroll
      for (int s = 0; s < 16; ++s) {
        int kk = kc * 32 + kmap(s >> 1, g) + (s & 1);
        b0[s] = Kt[kk][l16];
        b1[s] = Kt[kk][16 + l16];
      }
      s0 = wmma16x16x32(qf[kc], b0, s0);
      s1 = wmma16x16x32(qf[kc], b1, s1);
    }

    // online softmax; C layout: row = v + 8g, col = l16
#pragma unroll
    for (int v = 0; v < 8; ++v) {
      float e0 = s0[v] * scale, e1 = s1[v] * scale;
      float lmax = fmaxf(e0, e1);
#pragma unroll
      for (int o = 8; o > 0; o >>= 1) lmax = fmaxf(lmax, __shfl_xor(lmax, o, 32));
      float newm = fmaxf(rowmax[v], lmax);
      float corr = __expf(rowmax[v] - newm);
      float p0 = __expf(e0 - newm), p1 = __expf(e1 - newm);
      float ps = p0 + p1;
#pragma unroll
      for (int o = 8; o > 0; o >>= 1) ps += __shfl_xor(ps, o, 32);
      rowsum[v] = rowsum[v] * corr + ps;
      rowmax[v] = newm;
#pragma unroll
      for (int n = 0; n < 8; ++n) o8[n][v] *= corr;
      Ps[wid][v + 8 * g][l16]      = (_Float16)p0;
      Ps[wid][v + 8 * g][16 + l16] = (_Float16)p1;
    }
    __asm__ __volatile__("" ::: "memory");  // keep LDS store->load order (same wave, DS in-order)

    // reload P in A-fragment layout, then out += P * V
    v16h pf;
#pragma unroll
    for (int v = 0; v < 8; ++v) {
      h2_t pp = *(const h2_t*)&Ps[wid][l16][kmap(v, g)];
      pf[2 * v] = pp[0]; pf[2 * v + 1] = pp[1];
    }
#pragma unroll
    for (int n = 0; n < 8; ++n) {
      v16h vf;
#pragma unroll
      for (int s = 0; s < 16; ++s)
        vf[s] = Vt[kmap(s >> 1, g) + (s & 1)][n * 16 + l16];
      o8[n] = wmma16x16x32(pf, vf, o8[n]);
    }
  }

  // write out: O[t][h*128 + d], normalized by rowsum
  int rowb = bq * 128 + wid * 16 + 8 * g;
#pragma unroll
  for (int n = 0; n < 8; ++n) {
    int col = h * HDIM + n * 16 + l16;
#pragma unroll
    for (int v = 0; v < 8; ++v) {
      float val = o8[n][v] / rowsum[v];
      O[(size_t)(rowb + v) * DIM + col] = (_Float16)val;
    }
  }
}

// ---------------------------------------------------------------------------
// Gated residual: out[t] = base[t] + gate * tmp[t]
// ---------------------------------------------------------------------------
__global__ __launch_bounds__(256)
void k_residual(const float* __restrict__ baseImg, const float* __restrict__ baseTxt,
                const float* __restrict__ tmp, const float* __restrict__ mod,
                int gate_idx, float* __restrict__ outImg, float* __restrict__ outTxt) {
  int t = blockIdx.x, tid = threadIdx.x;
  const float* gate = ((t < LT) ? (mod + 6 * DIM) : mod) + gate_idx * DIM;
  const float* b = (t < LT) ? (baseTxt + (size_t)t * DIM)
                            : (baseImg + (size_t)(t - LT) * DIM);
  float* o = (t < LT) ? (outTxt + (size_t)t * DIM)
                      : (outImg + (size_t)(t - LT) * DIM);
  const float* tp = tmp + (size_t)t * DIM;
#pragma unroll
  for (int i = 0; i < 8; ++i) {
    int c = tid + i * 256;
    o[c] = b[c] + gate[c] * tp[c];
  }
}

// ---------------------------------------------------------------------------
// Workspace layout (bytes)
// ---------------------------------------------------------------------------
static constexpr size_t OFF_MOD  = 0;
static constexpr size_t SZ_MOD   = 2ull * 6 * DIM * 4;          //   98304
static constexpr size_t OFF_XMOD = OFF_MOD + SZ_MOD;
static constexpr size_t SZ_XMOD  = (size_t)LTOT * DIM * 2;      // 9437184
static constexpr size_t OFF_QKV  = OFF_XMOD + SZ_XMOD;
static constexpr size_t SZ_QKV   = (size_t)LTOT * 3 * DIM * 4;  // 56.6 MB
static constexpr size_t OFF_Q    = OFF_QKV + SZ_QKV;
static constexpr size_t SZ_HEAD  = (size_t)NH * LTOT * HDIM * 2;
static constexpr size_t OFF_K    = OFF_Q + SZ_HEAD;
static constexpr size_t OFF_V    = OFF_K + SZ_HEAD;
static constexpr size_t OFF_AO   = OFF_V + SZ_HEAD;
static constexpr size_t SZ_AO    = (size_t)LTOT * DIM * 2;
static constexpr size_t OFF_TMP  = OFF_AO + SZ_AO;
static constexpr size_t SZ_TMP   = (size_t)LTOT * DIM * 4;
static constexpr size_t OFF_HMID = OFF_TMP + SZ_TMP;

extern "C" void kernel_launch(void* const* d_in, const int* in_sizes, int n_in,
                              void* d_out, int out_size, void* d_ws, size_t ws_size,
                              hipStream_t stream) {
  (void)in_sizes; (void)n_in; (void)out_size; (void)ws_size;
  const float* img = (const float*)d_in[0];
  const float* txt = (const float*)d_in[1];
  const float* vec = (const float*)d_in[2];
  const float* pe  = (const float*)d_in[3];
  const float* i_mod_w  = (const float*)d_in[4];
  const float* i_mod_b  = (const float*)d_in[5];
  const float* i_qkv_w  = (const float*)d_in[6];
  const float* i_qs     = (const float*)d_in[7];
  const float* i_ks     = (const float*)d_in[8];
  const float* i_proj_w = (const float*)d_in[9];
  const float* i_proj_b = (const float*)d_in[10];
  const float* i_w1     = (const float*)d_in[11];
  const float* i_b1     = (const float*)d_in[12];
  const float* i_w2     = (const float*)d_in[13];
  const float* i_b2     = (const float*)d_in[14];
  const float* t_mod_w  = (const float*)d_in[15];
  const float* t_mod_b  = (const float*)d_in[16];
  const float* t_qkv_w  = (const float*)d_in[17];
  const float* t_qs     = (const float*)d_in[18];
  const float* t_ks     = (const float*)d_in[19];
  const float* t_proj_w = (const float*)d_in[20];
  const float* t_proj_b = (const float*)d_in[21];
  const float* t_w1     = (const float*)d_in[22];
  const float* t_b1     = (const float*)d_in[23];
  const float* t_w2     = (const float*)d_in[24];
  const float* t_b2     = (const float*)d_in[25];

  char* ws = (char*)d_ws;
  float*     mod   = (float*)(ws + OFF_MOD);
  _Float16*  xmod  = (_Float16*)(ws + OFF_XMOD);
  float*     qkv   = (float*)(ws + OFF_QKV);
  _Float16*  Qh    = (_Float16*)(ws + OFF_Q);
  _Float16*  Kh    = (_Float16*)(ws + OFF_K);
  _Float16*  Vh    = (_Float16*)(ws + OFF_V);
  _Float16*  attno = (_Float16*)(ws + OFF_AO);
  float*     tmp   = (float*)(ws + OFF_TMP);
  _Float16*  hmid  = (_Float16*)(ws + OFF_HMID);

  float* outImg = (float*)d_out;              // [LI][DIM]
  float* outTxt = outImg + (size_t)LI * DIM;  // [LT][DIM]

  // 0) modulation vectors
  k_modulation<<<96, 256, 0, stream>>>(vec, i_mod_w, i_mod_b, t_mod_w, t_mod_b, mod);

  // 1) LN + (sh1, sc1) -> xmod (f16)
  k_ln_mod<<<LTOT, 256, 0, stream>>>(img, txt, mod, 0, 1, xmod);

  // 2) QKV GEMMs (txt rows 0..255, img rows 256..)
  k_gemm<0,0,0><<<dim3(3*DIM/128, LT/128), 256, 0, stream>>>(
      xmod, DIM, t_qkv_w, 3*DIM, nullptr, qkv, 3*DIM, LT, 3*DIM, DIM);
  k_gemm<0,0,0><<<dim3(3*DIM/128, LI/128), 256, 0, stream>>>(
      xmod + (size_t)LT*DIM, DIM, i_qkv_w, 3*DIM, nullptr,
      qkv + (size_t)LT*3*DIM, 3*DIM, LI, 3*DIM, DIM);

  // 3) RMSNorm + RoPE -> head-major f16 Q/K/V
  k_qknorm_rope<<<dim3(NH, LTOT), 64, 0, stream>>>(qkv, pe, i_qs, i_ks, t_qs, t_ks,
                                                   Qh, Kh, Vh);

  // 4) flash attention -> attno (f16, [t][h*128+d])
  k_attn<<<dim3(NH, LTOT/128), 256, 0, stream>>>(Qh, Kh, Vh, attno);

  // 5) proj GEMMs (+bias) -> tmp (f32), then gated residual -> d_out
  k_gemm<0,1,0><<<dim3(DIM/128, LT/128), 256, 0, stream>>>(
      attno, DIM, t_proj_w, DIM, t_proj_b, tmp, DIM, LT, DIM, DIM);
  k_gemm<0,1,0><<<dim3(DIM/128, LI/128), 256, 0, stream>>>(
      attno + (size_t)LT*DIM, DIM, i_proj_w, DIM, i_proj_b,
      tmp + (size_t)LT*DIM, DIM, LI, DIM, DIM);
  k_residual<<<LTOT, 256, 0, stream>>>(img, txt, tmp, mod, 2, outImg, outTxt);

  // 6) LN2 + (sh2, sc2) -> xmod (f16)
  k_ln_mod<<<LTOT, 256, 0, stream>>>(outImg, outTxt, mod, 3, 4, xmod);

  // 7) MLP1 (+bias, GELU) -> hmid (f16)
  k_gemm<1,1,1><<<dim3(MLPD/128, LT/128), 256, 0, stream>>>(
      xmod, DIM, t_w1, MLPD, t_b1, hmid, MLPD, LT, MLPD, DIM);
  k_gemm<1,1,1><<<dim3(MLPD/128, LI/128), 256, 0, stream>>>(
      xmod + (size_t)LT*DIM, DIM, i_w1, MLPD, i_b1,
      hmid + (size_t)LT*MLPD, MLPD, LI, MLPD, DIM);

  // 8) MLP2 (+bias) -> tmp, gated residual in-place on d_out
  k_gemm<0,1,0><<<dim3(DIM/128, LT/128), 256, 0, stream>>>(
      hmid, MLPD, t_w2, DIM, t_b2, tmp, DIM, LT, DIM, MLPD);
  k_gemm<0,1,0><<<dim3(DIM/128, LI/128), 256, 0, stream>>>(
      hmid + (size_t)LT*MLPD, MLPD, i_w2, DIM, i_b2,
      tmp + (size_t)LT*DIM, DIM, LI, DIM, MLPD);
  k_residual<<<LTOT, 256, 0, stream>>>(outImg, outTxt, tmp, mod, 5, outImg, outTxt);
}
